// MultiheadSelfAttention_37203006718417
// MI455X (gfx1250) — compile-verified, matrix operand order checked by reading, not run on hardware
//
#include <hip/hip_runtime.h>
#include <hip/hip_bf16.h>

typedef __attribute__((ext_vector_type(16))) _Float16 v16h;
typedef __attribute__((ext_vector_type(8)))  _Float16 v8h;
typedef __attribute__((ext_vector_type(8)))  float    v8f;
typedef unsigned int uint32x4 __attribute__((ext_vector_type(4)));
typedef int          int32x8  __attribute__((ext_vector_type(8)));
typedef int          int32x4  __attribute__((ext_vector_type(4)));

#define DMODEL 1024
#define NHEADS 16
#define DK     64
#define SEQ    2048
#define BATCH  2

// ---------------------------------------------------------------------------
// Fragment loader: A-layout 16x32 f16 fragment (also B-layout when memory
// holds the operand row-major N x K). Lane L (r = L&15, h = L>>4):
//   v0..v3 <- row r, cols h*8 .. h*8+7
//   v4..v7 <- row r, cols 16+h*8 .. 16+h*8+7
// ---------------------------------------------------------------------------
__device__ __forceinline__ v16h load_frag(const _Float16* p, int stride) {
  int lane = threadIdx.x & 31;
  int r = lane & 15;
  int h = lane >> 4;
  const _Float16* q = p + (size_t)r * stride + h * 8;
  union { v8h half8[2]; v16h v; } u;
  u.half8[0] = *(const v8h*)(q);
  u.half8[1] = *(const v8h*)(q + 16);
  return u.v;
}

__device__ __forceinline__ v8f wmma32(v16h a, v16h b, v8f c) {
  return __builtin_amdgcn_wmma_f32_16x16x32_f16(false, a, false, b,
                                                (short)0, c, false, false);
}

// ---------------------------------------------------------------------------
// TDM: build a D# (ISA 8.3/8.4) for a 2-D f16 tile load Global -> LDS.
//   tile = tile_rows x tile_cols (f16), source rows are stride0 elems apart.
//   LDS result is packed row-major tile_rows x tile_cols.
// clang-23 toolchain: 6-arg builtin (g0, g1, g2, g3, extra, cpol).
// ---------------------------------------------------------------------------
__device__ __forceinline__ void tdm_load_2d(unsigned lds_off, const void* gptr,
                                            unsigned tensor_d0, unsigned tensor_d1,
                                            unsigned stride0,
                                            unsigned tile_cols, unsigned tile_rows) {
  unsigned long long ga = (unsigned long long)(uintptr_t)gptr;
  uint32x4 g0;
  g0.x = 1u;                                                // count=1 (valid)
  g0.y = lds_off;                                           // lds_addr [63:32]
  g0.z = (unsigned)ga;                                      // global_addr[31:0]
  g0.w = (unsigned)((ga >> 32) & 0x01FFFFFFu) | (2u << 30); // ga[56:32] | type=2
  int32x8 g1;
  g1[0] = (int)(1u << 16);                                  // data_size=1 -> 2 bytes
  g1[1] = (int)((tensor_d0 & 0xFFFFu) << 16);               // dim0[15:0] @ [63:48]
  g1[2] = (int)((tensor_d0 >> 16) | ((tensor_d1 & 0xFFFFu) << 16));
  g1[3] = (int)((tensor_d1 >> 16) | (tile_cols << 16));     // tile_dim0
  g1[4] = (int)tile_rows;                                   // tile_dim1 (tile_dim2=0)
  g1[5] = (int)stride0;                                     // tensor_dim0_stride[31:0]
  g1[6] = 0;
  g1[7] = 0;
  int32x4 z4 = {0, 0, 0, 0};
  int32x8 z8 = {0, 0, 0, 0, 0, 0, 0, 0};
  __builtin_amdgcn_tensor_load_to_lds(g0, g1, z4, z4, z8, 0);
}

// ---------------------------------------------------------------------------
// fp32 -> f16 elementwise convert
// ---------------------------------------------------------------------------
__global__ void cvt_f32_f16(const float* __restrict__ in,
                            _Float16* __restrict__ out, int n) {
  int i = blockIdx.x * blockDim.x + threadIdx.x;
  if (i < n) out[i] = (_Float16)in[i];
}

// ---------------------------------------------------------------------------
// C(M,N) = A(MxK f16 rm) * B(NxK f16 rm)^T, TDM-staged, double-buffered LDS.
// Block: 256 thr = 8 waves (2 M x 4 N), block tile 128x128, wave tile 64x32.
// ---------------------------------------------------------------------------
template <bool HALF_OUT>
__global__ void gemm_f16_tdm(const _Float16* __restrict__ A,
                             const _Float16* __restrict__ Bm,
                             void* __restrict__ Cout,
                             int M, int N, int K) {
  __shared__ __align__(16) _Float16 sA[2][128 * 32];
  __shared__ __align__(16) _Float16 sB[2][128 * 32];
  int w    = threadIdx.x >> 5;
  int lane = threadIdx.x & 31;
  int wm = w >> 2, wn = w & 3;
  int mblock = blockIdx.y * 128;
  int nblock = blockIdx.x * 128;
  bool issuer = (threadIdx.x < 32);  // wave 0 drives the TDM (EXEC ignored)

  if (issuer) {
    tdm_load_2d((unsigned)(uintptr_t)&sA[0][0], A + (size_t)mblock * K,
                (unsigned)K, (unsigned)M, (unsigned)K, 32u, 128u);
    tdm_load_2d((unsigned)(uintptr_t)&sB[0][0], Bm + (size_t)nblock * K,
                (unsigned)K, (unsigned)N, (unsigned)K, 32u, 128u);
  }

  v8f c[4][2] = {};
  int nsteps = K / 32;
  for (int i = 0; i < nsteps; ++i) {
    int cur = i & 1;
    if (issuer) {
      if (i + 1 < nsteps) {
        const _Float16* ga = A + (size_t)mblock * K + (size_t)(i + 1) * 32;
        const _Float16* gb = Bm + (size_t)nblock * K + (size_t)(i + 1) * 32;
        tdm_load_2d((unsigned)(uintptr_t)&sA[cur ^ 1][0], ga,
                    (unsigned)K, (unsigned)M, (unsigned)K, 32u, 128u);
        tdm_load_2d((unsigned)(uintptr_t)&sB[cur ^ 1][0], gb,
                    (unsigned)K, (unsigned)N, (unsigned)K, 32u, 128u);
        __builtin_amdgcn_s_wait_tensorcnt((short)2);  // current pair complete
      } else {
        __builtin_amdgcn_s_wait_tensorcnt((short)0);
      }
    }
    __syncthreads();  // publish LDS tiles to all waves

    const _Float16* tA = &sA[cur][0] + (wm * 64) * 32;
    const _Float16* tB = &sB[cur][0] + (wn * 32) * 32;
    v16h af[4], bf[2];
#pragma unroll
    for (int f = 0; f < 4; ++f) af[f] = load_frag(tA + f * 16 * 32, 32);
#pragma unroll
    for (int f = 0; f < 2; ++f) bf[f] = load_frag(tB + f * 16 * 32, 32);
#pragma unroll
    for (int fm = 0; fm < 4; ++fm)
#pragma unroll
      for (int fn = 0; fn < 2; ++fn)
        c[fm][fn] = wmma32(af[fm], bf[fn], c[fm][fn]);
    __syncthreads();  // reads done before next TDM overwrite
  }

  int n = lane & 15, hh = lane >> 4;
#pragma unroll
  for (int fm = 0; fm < 4; ++fm)
#pragma unroll
    for (int fn = 0; fn < 2; ++fn) {
      int mb = mblock + wm * 64 + fm * 16;
      int nb = nblock + wn * 32 + fn * 16;
#pragma unroll
      for (int r = 0; r < 8; ++r) {
        size_t idx = (size_t)(mb + r + hh * 8) * N + (nb + n);
        if (HALF_OUT) ((_Float16*)Cout)[idx] = (_Float16)c[fm][fn][r];
        else          ((float*)Cout)[idx]    = c[fm][fn][r];
      }
    }
}

// ---------------------------------------------------------------------------
// RoPE (fp32 trig) + layout change:
//   Q,K: (b*T, 1024) -> (b, h, t, dk)
//   V  : (b*T, 1024) -> (b, h, dk, t)   (transposed for B-fragment loads)
// ---------------------------------------------------------------------------
__global__ void rope_reorder(const _Float16* __restrict__ Qt,
                             const _Float16* __restrict__ Kt,
                             const _Float16* __restrict__ Vt,
                             const int* __restrict__ tok,
                             _Float16* __restrict__ Q,
                             _Float16* __restrict__ K,
                             _Float16* __restrict__ V) {
  int t = blockIdx.x, h = blockIdx.y, b = blockIdx.z;
  int tid = threadIdx.x;  // 64 threads
  size_t src   = ((size_t)(b * SEQ + t)) * DMODEL + h * DK;
  size_t qkdst = (((size_t)(b * NHEADS + h)) * SEQ + t) * DK;
  float pos = (float)tok[t];

  if (tid < 32) {
    int j = tid;
    float inv = __expf((float)j * (-2.0f / (float)DK) * 9.210340371976184f);
    float ang = pos * inv;
    float cs = __cosf(ang), sn = __sinf(ang);
    float qe = (float)Qt[src + 2 * j], qo = (float)Qt[src + 2 * j + 1];
    Q[qkdst + 2 * j]     = (_Float16)(qe * cs - qo * sn);
    Q[qkdst + 2 * j + 1] = (_Float16)(qe * sn + qo * cs);
    float ke = (float)Kt[src + 2 * j], ko = (float)Kt[src + 2 * j + 1];
    K[qkdst + 2 * j]     = (_Float16)(ke * cs - ko * sn);
    K[qkdst + 2 * j + 1] = (_Float16)(ke * sn + ko * cs);
  }
  int d = tid;
  V[(((size_t)(b * NHEADS + h)) * DK + d) * SEQ + t] = Vt[src + d];
}

// ---------------------------------------------------------------------------
// Flash attention: wave owns 16 query rows x dk=64; 32-key steps.
// ---------------------------------------------------------------------------
__global__ void attn_flash(const _Float16* __restrict__ Q,
                           const _Float16* __restrict__ K,
                           const _Float16* __restrict__ V,
                           _Float16* __restrict__ O) {
  __shared__ __align__(16) _Float16 sP[8][16 * 32];
  int w    = threadIdx.x >> 5;
  int lane = threadIdx.x & 31;
  int n  = lane & 15;
  int hh = lane >> 4;
  int h = blockIdx.y, b = blockIdx.z;
  int qblock = blockIdx.x * 128;
  int qbase  = qblock + w * 16;

  const _Float16* Qh = Q + ((size_t)(b * NHEADS + h)) * SEQ * DK;
  const _Float16* Kh = K + ((size_t)(b * NHEADS + h)) * SEQ * DK;
  const _Float16* Vh = V + ((size_t)(b * NHEADS + h)) * DK * SEQ;

  v16h aq0 = load_frag(Qh + (size_t)qbase * DK, DK);
  v16h aq1 = load_frag(Qh + (size_t)qbase * DK + 32, DK);

  v8f acc[4] = {};
  float m_run[8], l_run[8];
#pragma unroll
  for (int r = 0; r < 8; ++r) { m_run[r] = -1e30f; l_run[r] = 0.0f; }
  const float scale = 0.125f;  // 1/sqrt(64)

  int kend = qblock + 128;
  if (kend > SEQ) kend = SEQ;  // block-uniform trip count

  for (int kt = 0; kt < kend; kt += 32) {
    v8f s01[2];
#pragma unroll
    for (int g = 0; g < 2; ++g) {
      v16h bk0 = load_frag(Kh + (size_t)(kt + g * 16) * DK, DK);
      v16h bk1 = load_frag(Kh + (size_t)(kt + g * 16) * DK + 32, DK);
      v8f s = {};
      s = wmma32(aq0, bk0, s);
      s = wmma32(aq1, bk1, s);
      s01[g] = s;
    }

    float p[2][8];
#pragma unroll
    for (int r = 0; r < 8; ++r) {
      int qi = qbase + r + hh * 8;
      float sv0 = s01[0][r] * scale;
      float sv1 = s01[1][r] * scale;
      bool msk0 = (kt + n) > qi;
      bool msk1 = (kt + 16 + n) > qi;
      if (msk0) sv0 = -1e30f;
      if (msk1) sv1 = -1e30f;
      float mx = fmaxf(sv0, sv1);
#pragma unroll
      for (int off = 1; off < 16; off <<= 1)
        mx = fmaxf(mx, __shfl_xor(mx, off, 32));
      float m_new = fmaxf(m_run[r], mx);
      float alpha = __expf(m_run[r] - m_new);
      float e0 = msk0 ? 0.0f : __expf(sv0 - m_new);
      float e1 = msk1 ? 0.0f : __expf(sv1 - m_new);
      p[0][r] = e0;
      p[1][r] = e1;
      float rs = e0 + e1;
#pragma unroll
      for (int off = 1; off < 16; off <<= 1)
        rs += __shfl_xor(rs, off, 32);
      l_run[r] = l_run[r] * alpha + rs;
      m_run[r] = m_new;
#pragma unroll
      for (int g = 0; g < 4; ++g) acc[g][r] *= alpha;
    }

    __syncthreads();
    _Float16* tile = sP[w];
#pragma unroll
    for (int r = 0; r < 8; ++r) {
      int row = r + hh * 8;
      tile[row * 32 + n]      = (_Float16)p[0][r];
      tile[row * 32 + 16 + n] = (_Float16)p[1][r];
    }
    __syncthreads();
    v16h ap = load_frag(tile, 32);

#pragma unroll
    for (int g = 0; g < 4; ++g) {
      v16h bv = load_frag(Vh + (size_t)(g * 16) * SEQ + kt, SEQ);
      acc[g] = wmma32(ap, bv, acc[g]);
    }
  }

#pragma unroll
  for (int r = 0; r < 8; ++r) {
    float inv = (l_run[r] > 0.0f) ? 1.0f / l_run[r] : 0.0f;
    int t = qbase + r + hh * 8;
    size_t base = ((size_t)(b * SEQ + t)) * DMODEL + h * DK;
#pragma unroll
    for (int g = 0; g < 4; ++g)
      O[base + g * 16 + n] = (_Float16)(acc[g][r] * inv);
  }
}

// ---------------------------------------------------------------------------
extern "C" void kernel_launch(void* const* d_in, const int* in_sizes, int n_in,
                              void* d_out, int out_size, void* d_ws, size_t ws_size,
                              hipStream_t stream) {
  const float* x  = (const float*)d_in[0];
  const float* Wq = (const float*)d_in[1];
  const float* Wk = (const float*)d_in[2];
  const float* Wv = (const float*)d_in[3];
  const float* Wo = (const float*)d_in[4];
  const int*  tok = (const int*)d_in[5];
  float* out = (float*)d_out;

  char* ws = (char*)d_ws;
  const size_t MB = 1u << 20;
  _Float16* xh  = (_Float16*)(ws + 0 * MB);   // 8 MiB  (4096x1024)
  _Float16* Wqh = (_Float16*)(ws + 8 * MB);   // 2 MiB each
  _Float16* Wkh = (_Float16*)(ws + 10 * MB);
  _Float16* Wvh = (_Float16*)(ws + 12 * MB);
  _Float16* Woh = (_Float16*)(ws + 14 * MB);
  _Float16* Qt  = (_Float16*)(ws + 16 * MB);  // 8 MiB proj outputs
  _Float16* Kt  = (_Float16*)(ws + 24 * MB);
  _Float16* Vt  = (_Float16*)(ws + 32 * MB);
  _Float16* Qr  = (_Float16*)(ws + 40 * MB);  // (b,h,t,dk)
  _Float16* Kr  = (_Float16*)(ws + 48 * MB);
  _Float16* Vr  = (_Float16*)(ws + 56 * MB);  // (b,h,dk,t)
  _Float16* Ob  = Qt;                         // reuse Qt region for O

  const int M = BATCH * SEQ;  // 4096

  cvt_f32_f16<<<(M * DMODEL + 255) / 256, 256, 0, stream>>>(x, xh, M * DMODEL);
  cvt_f32_f16<<<(DMODEL * DMODEL + 255) / 256, 256, 0, stream>>>(Wq, Wqh, DMODEL * DMODEL);
  cvt_f32_f16<<<(DMODEL * DMODEL + 255) / 256, 256, 0, stream>>>(Wk, Wkh, DMODEL * DMODEL);
  cvt_f32_f16<<<(DMODEL * DMODEL + 255) / 256, 256, 0, stream>>>(Wv, Wvh, DMODEL * DMODEL);
  cvt_f32_f16<<<(DMODEL * DMODEL + 255) / 256, 256, 0, stream>>>(Wo, Woh, DMODEL * DMODEL);

  dim3 ggrid(DMODEL / 128, M / 128);
  gemm_f16_tdm<true><<<ggrid, 256, 0, stream>>>(xh, Wqh, Qt, M, DMODEL, DMODEL);
  gemm_f16_tdm<true><<<ggrid, 256, 0, stream>>>(xh, Wkh, Kt, M, DMODEL, DMODEL);
  gemm_f16_tdm<true><<<ggrid, 256, 0, stream>>>(xh, Wvh, Vt, M, DMODEL, DMODEL);

  rope_reorder<<<dim3(SEQ, NHEADS, BATCH), 64, 0, stream>>>(Qt, Kt, Vt, tok,
                                                            Qr, Kr, Vr);

  attn_flash<<<dim3(SEQ / 128, NHEADS, BATCH), 256, 0, stream>>>(Qr, Kr, Vr, Ob);

  gemm_f16_tdm<false><<<ggrid, 256, 0, stream>>>(Ob, Woh, out, M, DMODEL, DMODEL);
}
